// upconv_layer_batch_17557826306187
// MI455X (gfx1250) — matrix operand order repaired and checked by reading, not compiled, not run on hardware
//
#include <hip/hip_runtime.h>

// ---------------- problem constants ----------------
#define NN      10242          // N
#define N_PAD   10304          // 161 * 64, padded column count for GEMM tiles
#define CINN    256
#define COUTN   128
#define MROWS   896            // 7 * COUT
#define NBATCH  16
#define ZCOLS   (7 * NN)       // 71694
#define MOUT    40962          // 4*N - 6

typedef unsigned int  u32;
typedef unsigned short u16;
typedef __attribute__((ext_vector_type(2)))  float  f32x2;
typedef __attribute__((ext_vector_type(4)))  float  f32x4;
typedef __attribute__((ext_vector_type(4)))  u32    u32x4;
typedef __attribute__((ext_vector_type(8)))  float  v8f;
typedef __attribute__((ext_vector_type(16))) __bf16 v16bf;

union ABits { u32x4 u[2]; v16bf v; };

__device__ __forceinline__ u16 f2bf(float f) {
    __bf16 h = (__bf16)f;                      // RNE convert
    return __builtin_bit_cast(u16, h);
}

// ---------------- kernel 1: W f32 -> bf16 ----------------
__global__ __launch_bounds__(256) void cvtw_kernel(const float* __restrict__ W,
                                                   u16* __restrict__ wb) {
    int idx = blockIdx.x * 256 + threadIdx.x;   // grid = 896 blocks -> exactly 896*256 elems
    wb[idx] = f2bf(W[idx]);
}

// ---------------- kernel 2: x (B,CIN,N) f32 -> xbT (B,N_PAD,CIN) bf16 ----------------
// Tiled transpose through LDS: reads coalesced along n, writes coalesced along c.
__global__ __launch_bounds__(256) void cvtx_kernel(const float* __restrict__ x,
                                                   u16* __restrict__ xbT) {
    __shared__ u16 lds[256 * 33];              // [c][n] tile, padded stride 33
    const int b    = blockIdx.y;
    const int n0   = blockIdx.x * 32;
    const int c    = threadIdx.x;              // 0..255
    const int nrem = (NN - n0) < 32 ? (NN - n0) : 32;

    const float* src = x + ((size_t)b * CINN + c) * NN + n0;
    if (nrem == 32) {
        #pragma unroll
        for (int q = 0; q < 16; ++q) {         // 16 x f32x2 (rows are 8B-aligned)
            f32x2 v = *(const f32x2*)(src + 2 * q);
            lds[c * 33 + 2 * q]     = f2bf(v.x);
            lds[c * 33 + 2 * q + 1] = f2bf(v.y);
        }
    } else {
        for (int i = 0; i < nrem; ++i) lds[c * 33 + i] = f2bf(src[i]);
    }
    __syncthreads();

    const int n = threadIdx.x >> 3;            // 0..31
    const int p = threadIdx.x & 7;             // 0..7  -> c chunk of 32
    if (n < nrem) {
        u16* dst = xbT + ((size_t)b * N_PAD + n0 + n) * CINN + p * 32;
        #pragma unroll
        for (int q = 0; q < 4; ++q) {          // 4 x 16B stores = 32 bf16
            u32x4 u;
            #pragma unroll
            for (int w = 0; w < 4; ++w) {
                u16 lo = lds[(p * 32 + q * 8 + 2 * w)     * 33 + n];
                u16 hi = lds[(p * 32 + q * 8 + 2 * w + 1) * 33 + n];
                u[w] = (u32)lo | ((u32)hi << 16);
            }
            *(u32x4*)(dst + q * 8) = u;
        }
    }
}

// ---------------- kernel 3: bf16 WMMA GEMM for one batch ----------------
// zT[p][co] = W[j*128+co,:] . x[:,n] + bias,  p = n*7 + j
// grid = (161 n-tiles, 7 j), block = 256 (8 waves), wave tile = 16(M) x 64(N)
__global__ __launch_bounds__(256) void gemm_kernel(const u16* __restrict__ xbT_b,
                                                   const u16* __restrict__ wb,
                                                   const float* __restrict__ bias,
                                                   float* __restrict__ zT) {
    const int lane = threadIdx.x & 31;
    const int wave = threadIdx.x >> 5;         // 0..7 -> co0 = wave*16
    const int g    = lane >> 4;                // half-wave group 0/1
    const int lm   = lane & 15;
    const int j    = blockIdx.y;               // 0..6
    const int n0   = blockIdx.x * 64;
    const int co0  = wave * 16;

    const u16* aRow = wb + (size_t)(j * 128 + co0 + lm) * CINN;

    v8f acc[4] = {v8f(0.f), v8f(0.f), v8f(0.f), v8f(0.f)};

    #pragma unroll
    for (int k0 = 0; k0 < CINN; k0 += 32) {
        // A fragment: lane holds row (co0+lm), 16 contiguous K at k0+g*16
        ABits a;
        a.u[0] = *(const u32x4*)(aRow + k0 + g * 16);
        a.u[1] = *(const u32x4*)(aRow + k0 + g * 16 + 8);
        #pragma unroll
        for (int nt = 0; nt < 4; ++nt) {
            // B fragment: lane holds column (n0+nt*16+lm), 16 contiguous K at k0+g*16
            const u16* bp = xbT_b + (size_t)(n0 + nt * 16 + lm) * CINN + k0 + g * 16;
            v16bf bv = *(const v16bf*)bp;      // 32B aligned contiguous load
            acc[nt] = __builtin_amdgcn_wmma_f32_16x16x32_bf16(
                false, a.v, false, bv, (short)0, acc[nt], false, false);
        }
    }

    // bias: lane's 8 output rows are co0 + g*8 + r  (contiguous)
    const float* bp = bias + j * 128 + co0 + g * 8;
    f32x4 b0 = *(const f32x4*)bp;
    f32x4 b1 = *(const f32x4*)(bp + 4);

    #pragma unroll
    for (int nt = 0; nt < 4; ++nt) {
        int nIdx = n0 + nt * 16 + lm;
        if (nIdx < NN) {
            size_t p = (size_t)nIdx * 7 + j;
            float* o = zT + p * COUTN + co0 + g * 8;
            f32x4 lo, hi;
            lo.x = acc[nt][0] + b0.x; lo.y = acc[nt][1] + b0.y;
            lo.z = acc[nt][2] + b0.z; lo.w = acc[nt][3] + b0.w;
            hi.x = acc[nt][4] + b1.x; hi.y = acc[nt][5] + b1.y;
            hi.z = acc[nt][6] + b1.z; hi.w = acc[nt][7] + b1.w;
            *(f32x4*)o       = lo;             // 2 x 16B contiguous stores per lane
            *(f32x4*)(o + 4) = hi;
        }
    }
}

// ---------------- kernel 4: gather + edge-mean for one batch ----------------
// Reads 512B-contiguous columns of zT (L2-hot), transposes via LDS, writes out coalesced.
__global__ __launch_bounds__(256) void gather_kernel(const float* __restrict__ zT,
                                                     const int* __restrict__ cidx,
                                                     const int* __restrict__ eidx,
                                                     float* __restrict__ outB) {
    __shared__ float lds[128 * 33];            // [co][mi], padded
    const int t    = threadIdx.x;
    const int m0   = blockIdx.x * 32;
    const int mi   = t >> 3;                   // 0..31 output column within tile
    const int part = t & 7;                    // co chunk of 16
    const int cb   = part * 16;
    const int m    = m0 + mi;

    if (m < MOUT) {
        union { f32x4 v4[4]; float f[16]; } vv;
        if (m < NN) {
            int p = cidx[m];
            const f32x4* s = (const f32x4*)(zT + (size_t)p * COUTN + cb);
            #pragma unroll
            for (int q = 0; q < 4; ++q) vv.v4[q] = s[q];
        } else {
            int e = m - NN;
            int p0 = eidx[2 * e], p1 = eidx[2 * e + 1];
            const f32x4* s0 = (const f32x4*)(zT + (size_t)p0 * COUTN + cb);
            const f32x4* s1 = (const f32x4*)(zT + (size_t)p1 * COUTN + cb);
            #pragma unroll
            for (int q = 0; q < 4; ++q) vv.v4[q] = 0.5f * (s0[q] + s1[q]);
        }
        #pragma unroll
        for (int i = 0; i < 16; ++i) lds[(cb + i) * 33 + mi] = vv.f[i];
    }
    __syncthreads();

    const int co = t >> 1;                     // 0..127
    const int h  = t & 1;                      // halves of the 32 columns
    #pragma unroll
    for (int i = 0; i < 16; ++i) {
        int m2 = m0 + h * 16 + i;
        if (m2 < MOUT)
            outB[(size_t)co * MOUT + m2] = lds[co * 33 + h * 16 + i];
    }
}

// ---------------- workspace layout (bytes) ----------------
static constexpr size_t XBT_BYTES = (size_t)NBATCH * N_PAD * CINN * 2;  // 84,410,368
static constexpr size_t WB_OFF    = XBT_BYTES;                          // 256B aligned
static constexpr size_t WB_BYTES  = (size_t)MROWS * CINN * 2;           // 458,752
static constexpr size_t ZT_OFF    = WB_OFF + WB_BYTES;                  // 16B aligned

extern "C" void kernel_launch(void* const* d_in, const int* in_sizes, int n_in,
                              void* d_out, int out_size, void* d_ws, size_t ws_size,
                              hipStream_t stream) {
    const float* x    = (const float*)d_in[0];
    const float* W    = (const float*)d_in[1];
    const float* bias = (const float*)d_in[2];
    const int*   cidx = (const int*)d_in[3];
    const int*   eidx = (const int*)d_in[4];
    float*       out  = (float*)d_out;
    char*        ws   = (char*)d_ws;

    u16*   xbT = (u16*)(ws);
    u16*   wb  = (u16*)(ws + WB_OFF);
    float* zT  = (float*)(ws + ZT_OFF);

    cvtw_kernel<<<MROWS, 256, 0, stream>>>(W, wb);
    cvtx_kernel<<<dim3((NN + 31) / 32, NBATCH), 256, 0, stream>>>(x, xbT);

    // batch-by-batch: zT (36.7MB) stays L2-resident between GEMM and gather
    for (int b = 0; b < NBATCH; ++b) {
        gemm_kernel<<<dim3(N_PAD / 64, 7), 256, 0, stream>>>(
            xbT + (size_t)b * N_PAD * CINN, wb, bias, zT);
        gather_kernel<<<(MOUT + 31) / 32, 256, 0, stream>>>(
            zT, cidx, eidx, out + (size_t)b * COUTN * MOUT);
    }
}
